// GCN_fix_2layer_64hidden_80118319939769
// MI455X (gfx1250) — compile-verified, matrix operand order checked by reading, not run on hardware
//
#include <hip/hip_runtime.h>
#include <hip/hip_bf16.h>

typedef __attribute__((ext_vector_type(16))) __bf16 v16bf;
typedef __attribute__((ext_vector_type(8)))  float  v8f;

#define FH 64   // hidden width (fixed by model)

// ---------------- degree / normalization ----------------
__global__ void k_init_deg(float* deg, int N) {
    int i = blockIdx.x * blockDim.x + threadIdx.x;
    if (i < N) deg[i] = 1.0f;                      // self-loop contributes 1
}

__global__ void k_deg_accum(const long long* __restrict__ ei, float* deg, int E) {
    int e = blockIdx.x * blockDim.x + threadIdx.x;
    if (e < E) atomicAdd(&deg[ei[(size_t)E + e]], 1.0f);   // dst half
}

__global__ void k_deg_finalize(float* deg, int N) {
    int i = blockIdx.x * blockDim.x + threadIdx.x;
    if (i < N) deg[i] = rsqrtf(fmaxf(deg[i], 1.0f));       // in-place -> deg_inv_sqrt
}

__global__ void k_zero(float* p, size_t n) {
    size_t i = (size_t)blockIdx.x * blockDim.x + threadIdx.x;
    if (i < n) p[i] = 0.0f;
}

// ---------------- W1 -> bf16 B-fragments, pre-swizzled to WMMA layout --------
// Layout: fragment16[(tile*nks + ks)*32 + lane] : 16 bf16 = one lane's B slice.
__global__ void k_prep_w1(const float* __restrict__ W1, __bf16* __restrict__ wb,
                          int K, int nks) {
    const int idx = blockIdx.x * blockDim.x + threadIdx.x;
    const int total = 4 * nks * 32;
    if (idx >= total) return;
    const int lane = idx & 31;
    const int tmp  = idx >> 5;
    const int ks   = tmp % nks;
    const int tile = tmp / nks;
    const int n    = lane & 15;
    const int half = lane >> 4;
    const int col  = tile * 16 + n;
    const int ka   = ks * 32 + half * 8;
    __bf16* o = wb + (size_t)idx * 16;
#pragma unroll
    for (int t = 0; t < 8; ++t) {
        const int k1 = ka + t, k2 = ka + 16 + t;
        o[t]     = (k1 < K) ? (__bf16)W1[(size_t)k1 * FH + col] : (__bf16)0.0f;
        o[8 + t] = (k2 < K) ? (__bf16)W1[(size_t)k2 * FH + col] : (__bf16)0.0f;
    }
}

// ---------------- GEMM1: h = x @ W1  (bf16 WMMA, f32 accum) ----------------
// Block = 128 threads = 4 waves; block covers 64 rows x 64 cols.
// Each wave owns 16 distinct rows and computes all four 16-col tiles,
// amortizing one A fragment over 4 WMMAs per k-step.
__global__ void __launch_bounds__(128)
k_gemm1_wmma(const float* __restrict__ x, const __bf16* __restrict__ wb,
             float* __restrict__ h, int N, int K, int nks) {
    const int wave = threadIdx.x >> 5;
    const int lane = threadIdx.x & 31;
    const int m    = lane & 15;       // row within wave tile / col within B tile
    const int half = lane >> 4;       // K-half selector for A/B fragments
    const int row0 = (blockIdx.x << 6) + (wave << 4);   // this wave's 16 rows
    const int row  = row0 + m;
    const int rowc = (row < N) ? row : (N - 1);   // clamp: loads always legal
    const float* __restrict__ xrow = x + (size_t)rowc * K;

    const v16bf* __restrict__ bp = (const v16bf*)wb + lane;   // + (c*nks+ks)*32

    const int koff  = half << 3;      // 0 or 8
    const int nfull = K >> 5;         // full 32-wide k-steps

    v8f acc0 = {}, acc1 = {}, acc2 = {}, acc3 = {};
    for (int ks = 0; ks < nfull; ++ks) {
        const int ka = (ks << 5) + koff;
        // ---- A fragment: 16 unconditional b32 loads -> one clause ----
        float af[16];
#pragma unroll
        for (int t = 0; t < 8; ++t) {
            af[t]     = xrow[ka + t];
            af[8 + t] = xrow[ka + 16 + t];
        }
        v16bf a;
#pragma unroll
        for (int t = 0; t < 16; ++t) a[t] = (__bf16)af[t];

        // ---- B fragments: four aligned 32-byte loads ----
        const v16bf b0 = bp[((size_t)0 * nks + ks) * 32];
        const v16bf b1 = bp[((size_t)1 * nks + ks) * 32];
        const v16bf b2 = bp[((size_t)2 * nks + ks) * 32];
        const v16bf b3 = bp[((size_t)3 * nks + ks) * 32];

        if (ka + 64 < K)
            __builtin_prefetch(xrow + ka + 64, 0, 1);   // global_prefetch_b8

        acc0 = __builtin_amdgcn_wmma_f32_16x16x32_bf16(false, a, false, b0, (short)0, acc0, false, false);
        acc1 = __builtin_amdgcn_wmma_f32_16x16x32_bf16(false, a, false, b1, (short)0, acc1, false, false);
        acc2 = __builtin_amdgcn_wmma_f32_16x16x32_bf16(false, a, false, b2, (short)0, acc2, false, false);
        acc3 = __builtin_amdgcn_wmma_f32_16x16x32_bf16(false, a, false, b3, (short)0, acc3, false, false);
    }

    // ---- K tail (K % 32 != 0): clamped addresses + selects, no branches ----
    if (K & 31) {
        const int ka = (nfull << 5) + koff;
        v16bf a;
#pragma unroll
        for (int t = 0; t < 8; ++t) {
            const int k1 = ka + t, k2 = ka + 16 + t;
            const float v1 = xrow[(k1 < K) ? k1 : (K - 1)];
            const float v2 = xrow[(k2 < K) ? k2 : (K - 1)];
            a[t]     = (k1 < K) ? (__bf16)v1 : (__bf16)0.0f;
            a[8 + t] = (k2 < K) ? (__bf16)v2 : (__bf16)0.0f;
        }
        const v16bf b0 = bp[((size_t)0 * nks + nfull) * 32];  // zero-padded rows
        const v16bf b1 = bp[((size_t)1 * nks + nfull) * 32];
        const v16bf b2 = bp[((size_t)2 * nks + nfull) * 32];
        const v16bf b3 = bp[((size_t)3 * nks + nfull) * 32];
        acc0 = __builtin_amdgcn_wmma_f32_16x16x32_bf16(false, a, false, b0, (short)0, acc0, false, false);
        acc1 = __builtin_amdgcn_wmma_f32_16x16x32_bf16(false, a, false, b1, (short)0, acc1, false, false);
        acc2 = __builtin_amdgcn_wmma_f32_16x16x32_bf16(false, a, false, b2, (short)0, acc2, false, false);
        acc3 = __builtin_amdgcn_wmma_f32_16x16x32_bf16(false, a, false, b3, (short)0, acc3, false, false);
    }

    // ---- store: C/D layout lanes 0-15 -> M=j, lanes 16-31 -> M=8+j; N=lane&15
    if (row0 + 16 <= N) {               // wave-uniform fast path: no guards
#pragma unroll
        for (int j = 0; j < 8; ++j) {
            const size_t base = (size_t)(row0 + j + (half << 3)) * FH + m;
            h[base]      = acc0[j];
            h[base + 16] = acc1[j];
            h[base + 32] = acc2[j];
            h[base + 48] = acc3[j];
        }
    } else {
#pragma unroll
        for (int j = 0; j < 8; ++j) {
            const int r = row0 + j + (half << 3);
            if (r < N) {
                const size_t base = (size_t)r * FH + m;
                h[base]      = acc0[j];
                h[base + 16] = acc1[j];
                h[base + 32] = acc2[j];
                h[base + 48] = acc3[j];
            }
        }
    }
}

// ---------------- edge aggregation, 64-wide: one wave per edge ----------------
__global__ void k_aggregate64(const long long* __restrict__ ei, const float* __restrict__ dis,
                              const float* __restrict__ h, float* __restrict__ agg, int E) {
    const size_t gt = (size_t)blockIdx.x * blockDim.x + threadIdx.x;
    const int e    = (int)(gt >> 5);
    const int lane = threadIdx.x & 31;
    if (e >= E) return;
    const long long s = ei[e];
    const long long d = ei[(size_t)E + e];
    const float norm = dis[s] * dis[d];
    const float* hs = h + (size_t)s * FH;
    float* od = agg + (size_t)d * FH;
    atomicAdd(&od[lane],      hs[lane]      * norm);
    atomicAdd(&od[lane + 32], hs[lane + 32] * norm);
}

// ---------------- finalize layer 1: add self-loop + bias, ReLU (in place) ----
__global__ void k_finalize1(float* agg, const float* __restrict__ h,
                            const float* __restrict__ dis, const float* __restrict__ b1, int N) {
    const size_t idx = (size_t)blockIdx.x * blockDim.x + threadIdx.x;
    if (idx >= (size_t)N * FH) return;
    const int i = (int)(idx >> 6);
    const int f = (int)(idx & 63);
    const float di = dis[i];
    const float v = agg[idx] + h[idx] * di * di + b1[f];
    agg[idx] = fmaxf(v, 0.0f);
}

// ---------------- GEMM2: h2 = z @ W2  (tiny: 64 -> Fo) ----------------
__global__ void k_gemm2(const float* __restrict__ z, const float* __restrict__ W2,
                        float* __restrict__ h2, int N, int Fo) {
    __shared__ float w[FH * 8];
    for (int i = threadIdx.x; i < FH * Fo; i += blockDim.x) w[i] = W2[i];
    __syncthreads();
    const int i = blockIdx.x * blockDim.x + threadIdx.x;
    if (i >= N) return;
    const float* zi = z + (size_t)i * FH;
    float acc[8];
#pragma unroll
    for (int o = 0; o < 8; ++o) acc[o] = 0.0f;
    for (int k = 0; k < FH; ++k) {
        const float zv = zi[k];
        for (int o = 0; o < Fo; ++o) acc[o] = fmaf(zv, w[k * Fo + o], acc[o]);
    }
    for (int o = 0; o < Fo; ++o) h2[(size_t)i * Fo + o] = acc[o];
}

// ---------------- edge aggregation, Fo-wide: thread per (edge, feature) ------
__global__ void k_aggregate7(const long long* __restrict__ ei, const float* __restrict__ dis,
                             const float* __restrict__ h2, float* __restrict__ out,
                             int E, int Fo) {
    const size_t idx = (size_t)blockIdx.x * blockDim.x + threadIdx.x;
    if (idx >= (size_t)E * Fo) return;
    const int e = (int)(idx / Fo);
    const int o = (int)(idx - (size_t)e * Fo);
    const long long s = ei[e];
    const long long d = ei[(size_t)E + e];
    const float norm = dis[s] * dis[d];
    atomicAdd(&out[(size_t)d * Fo + o], h2[(size_t)s * Fo + o] * norm);
}

// ---------------- finalize layer 2: self-loop + bias (no ReLU) ---------------
__global__ void k_finalize2(float* out, const float* __restrict__ h2,
                            const float* __restrict__ dis, const float* __restrict__ b2,
                            int N, int Fo) {
    const size_t idx = (size_t)blockIdx.x * blockDim.x + threadIdx.x;
    if (idx >= (size_t)N * Fo) return;
    const int i = (int)(idx / Fo);
    const int o = (int)(idx - (size_t)i * Fo);
    const float di = dis[i];
    out[idx] += h2[idx] * di * di + b2[o];
}

extern "C" void kernel_launch(void* const* d_in, const int* in_sizes, int n_in,
                              void* d_out, int out_size, void* d_ws, size_t ws_size,
                              hipStream_t stream) {
    const float*     x  = (const float*)d_in[0];
    const long long* ei = (const long long*)d_in[1];
    const float*     W1 = (const float*)d_in[2];
    const float*     b1 = (const float*)d_in[3];
    const float*     W2 = (const float*)d_in[4];
    const float*     b2 = (const float*)d_in[5];

    const int Fh = in_sizes[3];          // 64
    const int K  = in_sizes[2] / Fh;     // 1433
    const int N  = in_sizes[0] / K;      // 100000
    const int E  = in_sizes[1] / 2;      // 3200000
    const int Fo = in_sizes[5];          // 7
    const int nks = (K + 31) / 32;       // 45 k-steps (incl. zero-padded tail)
    (void)Fh; (void)n_in; (void)ws_size; (void)out_size;

    char* ws = (char*)d_ws;
    __bf16* w1bf = (__bf16*)ws;  ws += (((size_t)4 * nks * 32 * 16 * 2) + 255) & ~(size_t)255;
    float*  dis  = (float*)ws;   ws += (((size_t)N * 4)       + 255) & ~(size_t)255;
    float*  h1   = (float*)ws;   ws += (((size_t)N * FH * 4)  + 255) & ~(size_t)255;
    float*  agg1 = (float*)ws;   ws += (((size_t)N * FH * 4)  + 255) & ~(size_t)255;
    float*  h2   = (float*)ws;
    float*  out  = (float*)d_out;

    const int T = 256;

    // degrees -> deg_inv_sqrt
    k_init_deg    <<<(N + T - 1) / T, T, 0, stream>>>(dis, N);
    k_deg_accum   <<<(E + T - 1) / T, T, 0, stream>>>(ei, dis, E);
    k_deg_finalize<<<(N + T - 1) / T, T, 0, stream>>>(dis, N);

    // W1 -> pre-swizzled bf16 B fragments
    const int nprep = 4 * nks * 32;
    k_prep_w1<<<(nprep + T - 1) / T, T, 0, stream>>>(W1, w1bf, K, nks);

    // layer 1  (block = 64 rows x 64 cols)
    k_gemm1_wmma<<<(N + 63) / 64, 128, 0, stream>>>(x, w1bf, h1, N, K, nks);
    const size_t n1 = (size_t)N * FH;
    k_zero<<<(unsigned)((n1 + T - 1) / T), T, 0, stream>>>(agg1, n1);
    k_aggregate64<<<(unsigned)((((size_t)E * 32) + T - 1) / T), T, 0, stream>>>(ei, dis, h1, agg1, E);
    k_finalize1<<<(unsigned)((n1 + T - 1) / T), T, 0, stream>>>(agg1, h1, dis, b1, N);

    // layer 2
    k_gemm2<<<(N + T - 1) / T, T, 0, stream>>>(agg1, W2, h2, N, Fo);
    const size_t n2 = (size_t)N * Fo;
    k_zero<<<(unsigned)((n2 + T - 1) / T), T, 0, stream>>>(out, n2);
    k_aggregate7<<<(unsigned)((((size_t)E * Fo) + T - 1) / T), T, 0, stream>>>(ei, dis, h2, out, E, Fo);
    k_finalize2<<<(unsigned)((n2 + T - 1) / T), T, 0, stream>>>(out, h2, dis, b2, N, Fo);
}
